// NLBlock_10127532884658
// MI455X (gfx1250) — matrix operand order repaired and checked by reading, not compiled, not run on hardware
//
#include <hip/hip_runtime.h>
#include <hip/hip_bf16.h>
#include <math.h>

// ---------------------------------------------------------------------------
// NLBlock on MI455X (gfx1250, wave32).
//
// Algebraic collapse: k/v projections folded through the single-query
// attention so the big [B*T,D]x[D,D] GEMMs become [B,D]x[D,D] GEMMs:
//   qk  = (ct@W1^T + b1) @ W2          (q.b2 is softmax-invariant -> dropped)
//   sl  = softmax( scale * <qk[b], Lt[b,t,:]> )
//   u   = sum_t sl[t] * Lt[b,t,:]
//   sll = u @ W3^T + b3                (sum_t sl = 1 -> b3 exact)
//   out = ct + relu(LN(sll)) @ W4^T + b4
// HBM floor: one streaming read of Lt (268 MB) ~ 12 us @ 23.3 TB/s.
// GEMMs: v_wmma_f32_16x16x32_bf16, f32 accumulation.
// ---------------------------------------------------------------------------

#define DD 512
#define BB 2048
#define TT 64

typedef __attribute__((ext_vector_type(16))) __bf16 v16bf;
typedef __attribute__((ext_vector_type(8)))  float  v8f;

__device__ __forceinline__ __bf16 f2bf(float f) {
    // round-to-nearest-even f32 -> bf16, pure integer ops (toolchain-safe)
    unsigned u = __builtin_bit_cast(unsigned, f);
    unsigned short h = (unsigned short)((u + 0x7fffu + ((u >> 16) & 1u)) >> 16);
    return __builtin_bit_cast(__bf16, h);
}

// Y[m,n] = sum_k A[m,k] * B(k,n) (+bias[n]) (+resid[m,n])
// BT=true : B(k,n) = W[n*K + k]   (i.e. A @ W^T)
// BT=false: B(k,n) = W[k*N + n]   (i.e. A @ W)
template <bool BT, bool RES>
__global__ __launch_bounds__(128)
void gemm_wmma_bf16(const float* __restrict__ A, const float* __restrict__ W,
                    const float* __restrict__ bias, const float* __restrict__ resid,
                    float* __restrict__ Y)
{
    constexpr int M = BB, N = DD, K = DD;
    const int lane = threadIdx.x & 31;
    const int wave = threadIdx.x >> 5;          // 4 waves: tiles across N
    const int row0 = blockIdx.y * 16;           // M tile
    const int col0 = blockIdx.x * 64 + wave * 16;

    // A fragment addressing (contiguous k = 16*(lane/16) + s, same order as B)
    const int mA = lane & 15;
    const int kA = (lane >> 4) * 16;            // 0 or 16
    // B fragment addressing (16-bit B 32x16 layout)
    const int nB = lane & 15;
    const int kB = (lane >> 4) * 16;            // 0 or 16

    const float* Arow = A + (size_t)(row0 + mA) * K;

    v8f acc = {};
    for (int k0 = 0; k0 < K; k0 += 32) {
        v16bf a, b;
#pragma unroll
        for (int i = 0; i < 8; ++i) {
            a[i]     = f2bf(Arow[k0 + kA + i]);
            a[i + 8] = f2bf(Arow[k0 + kA + 8 + i]);
        }
        if (BT) {
            const float* Wr = W + (size_t)(col0 + nB) * K + (k0 + kB);
#pragma unroll
            for (int e = 0; e < 16; ++e) b[e] = f2bf(Wr[e]);
        } else {
            const float* Wc = W + (size_t)(k0 + kB) * N + (col0 + nB);
#pragma unroll
            for (int e = 0; e < 16; ++e) b[e] = f2bf(Wc[(size_t)e * N]);
        }
        acc = __builtin_amdgcn_wmma_f32_16x16x32_bf16(
                  false, a, false, b, (short)0, acc, false, false);
    }

    // C/D layout: element r -> M = r + (lane>=16 ? 8 : 0), N = lane&15
    const int nOut  = col0 + (lane & 15);
    const int mBase = row0 + ((lane >> 4) * 8);
    const float bv = bias ? bias[nOut] : 0.0f;
#pragma unroll
    for (int r = 0; r < 8; ++r) {
        float v = acc[r] + bv;
        if (RES) v += resid[(size_t)(mBase + r) * N + nOut];
        Y[(size_t)(mBase + r) * N + nOut] = v;
    }
}

// One block per batch element b: scores + softmax + weighted sum over T.
__global__ __launch_bounds__(256)
void attn_kernel(const float* __restrict__ qk, const float* __restrict__ Lt,
                 float* __restrict__ U)
{
    __shared__ float s_qk[DD];
    __shared__ float s_p[TT];

    const int b    = blockIdx.x;
    const int tid  = threadIdx.x;
    const int lane = tid & 31;
    const int wave = tid >> 5;                       // 8 waves
    const float scale = 0.04419417382415922f;        // 1/sqrt(512)

    const float* qkb = qk + (size_t)b * DD;
    const float* Lb  = Lt + (size_t)b * TT * DD;

    s_qk[tid]       = qkb[tid];
    s_qk[tid + 256] = qkb[tid + 256];
    __syncthreads();

    // scores: wave w handles t = 8w..8w+7; 32-lane dot over D
    for (int tt = 0; tt < 8; ++tt) {
        const int t = wave * 8 + tt;
        const float* row = Lb + (size_t)t * DD;
        if (t + 1 < TT) __builtin_prefetch(row + DD, 0, 0);
        float acc = 0.0f;
#pragma unroll
        for (int ii = 0; ii < DD / 32; ++ii) {
            const int i = lane + ii * 32;
            acc += s_qk[i] * row[i];
        }
#pragma unroll
        for (int off = 16; off > 0; off >>= 1)
            acc += __shfl_xor(acc, off, 32);
        if (lane == 0) s_p[t] = acc * scale;
    }
    __syncthreads();

    // softmax over T=64 done by wave 0 (2 values per lane)
    if (wave == 0) {
        float x0 = s_p[lane], x1 = s_p[lane + 32];
        float m = fmaxf(x0, x1);
#pragma unroll
        for (int off = 16; off > 0; off >>= 1)
            m = fmaxf(m, __shfl_xor(m, off, 32));
        float e0 = __expf(x0 - m), e1 = __expf(x1 - m);
        float s = e0 + e1;
#pragma unroll
        for (int off = 16; off > 0; off >>= 1)
            s += __shfl_xor(s, off, 32);
        float inv = 1.0f / s;
        s_p[lane]      = e0 * inv;
        s_p[lane + 32] = e1 * inv;
    }
    __syncthreads();

    // u[e] = sum_t p[t] * Lt[b,t,e]; coalesced column pass (L2 re-hit)
    float u0 = 0.0f, u1 = 0.0f;
    for (int t = 0; t < TT; ++t) {
        const float p = s_p[t];
        const float* row = Lb + (size_t)t * DD;
        u0 = fmaf(p, row[tid], u0);
        u1 = fmaf(p, row[tid + 256], u1);
    }
    U[(size_t)b * DD + tid]       = u0;
    U[(size_t)b * DD + tid + 256] = u1;
}

// LayerNorm over D=512 + ReLU, in place. One block per row.
__global__ __launch_bounds__(256)
void ln_relu_kernel(float* __restrict__ X, const float* __restrict__ g,
                    const float* __restrict__ bta)
{
    __shared__ float red[8];
    const int row = blockIdx.x, tid = threadIdx.x;
    const int lane = tid & 31, wave = tid >> 5;
    float* x = X + (size_t)row * DD;

    float v0 = x[tid], v1 = x[tid + 256];

    float s = v0 + v1;
#pragma unroll
    for (int off = 16; off > 0; off >>= 1) s += __shfl_xor(s, off, 32);
    if (lane == 0) red[wave] = s;
    __syncthreads();
    float tot = 0.0f;
#pragma unroll
    for (int i = 0; i < 8; ++i) tot += red[i];
    const float mu = tot * (1.0f / DD);
    __syncthreads();

    const float d0 = v0 - mu, d1 = v1 - mu;
    float sq = d0 * d0 + d1 * d1;
#pragma unroll
    for (int off = 16; off > 0; off >>= 1) sq += __shfl_xor(sq, off, 32);
    if (lane == 0) red[wave] = sq;
    __syncthreads();
    float tot2 = 0.0f;
#pragma unroll
    for (int i = 0; i < 8; ++i) tot2 += red[i];
    const float inv = rsqrtf(tot2 * (1.0f / DD) + 1e-5f);

    x[tid]       = fmaxf(0.0f, d0 * inv * g[tid] + bta[tid]);
    x[tid + 256] = fmaxf(0.0f, d1 * inv * g[tid + 256] + bta[tid + 256]);
}

extern "C" void kernel_launch(void* const* d_in, const int* in_sizes, int n_in,
                              void* d_out, int out_size, void* d_ws, size_t ws_size,
                              hipStream_t stream)
{
    (void)in_sizes; (void)n_in; (void)out_size; (void)ws_size;
    const float* ct   = (const float*)d_in[0];
    const float* Lt   = (const float*)d_in[1];
    const float* W1   = (const float*)d_in[2];
    const float* b1   = (const float*)d_in[3];
    const float* W2   = (const float*)d_in[4];
    /* b2 (d_in[5]) is softmax-invariant -> unused */
    const float* W3   = (const float*)d_in[6];
    const float* b3   = (const float*)d_in[7];
    const float* W4   = (const float*)d_in[8];
    const float* b4   = (const float*)d_in[9];
    const float* ln_g = (const float*)d_in[10];
    const float* ln_b = (const float*)d_in[11];
    float* out = (float*)d_out;

    float* q   = (float*)d_ws;                 // [B,D] 4 MB
    float* qk  = q  + (size_t)BB * DD;         // [B,D] 4 MB
    float* u   = qk + (size_t)BB * DD;         // [B,D] 4 MB
    float* sll = u  + (size_t)BB * DD;         // [B,D] 4 MB

    dim3 gg(DD / 64, BB / 16);                 // (8, 128), 4 waves/block

    // 1) q = ct @ W1^T + b1
    gemm_wmma_bf16<true,  false><<<gg, 128, 0, stream>>>(ct, W1, b1, nullptr, q);
    // 2) qk = q @ W2
    gemm_wmma_bf16<false, false><<<gg, 128, 0, stream>>>(q, W2, nullptr, nullptr, qk);
    // 3) scores + softmax + weighted sum over T (one HBM pass over Lt)
    attn_kernel<<<BB, 256, 0, stream>>>(qk, Lt, u);
    // 4) sll = u @ W3^T + b3
    gemm_wmma_bf16<true,  false><<<gg, 128, 0, stream>>>(u, W3, b3, nullptr, sll);
    // 5) LayerNorm + ReLU in place
    ln_relu_kernel<<<BB, 256, 0, stream>>>(sll, ln_g, ln_b);
    // 6) out = ct + sll @ W4^T + b4
    gemm_wmma_bf16<true,  true ><<<gg, 128, 0, stream>>>(sll, W4, b4, ct, out);
}
